// PatientGNN_10153302688286
// MI455X (gfx1250) — matrix operand order repaired and checked by reading, not compiled, not run on hardware
//
#include <hip/hip_runtime.h>

// MI455X / gfx1250, wave32, WMMA (not MFMA).
typedef _Float16 v16h __attribute__((ext_vector_type(16)));
typedef float    v8f  __attribute__((ext_vector_type(8)));

#define BN_EPS 1e-5f

static inline int cdiv_i(long long a, long long b) { return (int)((a + b - 1) / b); }

// ---------------- graph-norm kernels ----------------

__global__ void k_fill1(float* __restrict__ p, int n) {
    int i = blockIdx.x * blockDim.x + threadIdx.x;
    if (i < n) p[i] = 1.0f;   // self-loop contributes 1 to every degree
}

__global__ void k_deg(const int* __restrict__ dst, float* __restrict__ deg, int E) {
    int e = blockIdx.x * blockDim.x + threadIdx.x;
    if (e < E) atomicAdd(&deg[dst[e]], 1.0f);
}

__global__ void k_rsqrt_inplace(float* __restrict__ p, int n) {
    int i = blockIdx.x * blockDim.x + threadIdx.x;
    if (i < n) p[i] = rsqrtf(p[i]);   // deg >= 1 always (self-loop)
}

__global__ void k_norm(const int* __restrict__ src, const int* __restrict__ dst,
                       const float* __restrict__ dinv, float* __restrict__ norm, int E) {
    int e = blockIdx.x * blockDim.x + threadIdx.x;
    if (e < E) norm[e] = dinv[src[e]] * dinv[dst[e]];
}

__global__ void k_cvt_f32_f16(const float* __restrict__ s, _Float16* __restrict__ d, long long n) {
    long long i = (long long)blockIdx.x * blockDim.x + threadIdx.x;
    if (i < n) d[i] = (_Float16)s[i];
}

__global__ void k_zero4(float4* __restrict__ p, long long n4) {
    long long i      = (long long)blockIdx.x * blockDim.x + threadIdx.x;
    long long stride = (long long)gridDim.x * blockDim.x;
    float4 z; z.x = z.y = z.z = z.w = 0.0f;
    for (; i < n4; i += stride) p[i] = z;
}

// ---------------- WMMA GEMM: C[M x 128] = A16[M x 128] @ W16[128 x 128] ----------------
// Block = 256 threads = 8 waves. Wave w owns output columns [16w, 16w+16).
// B fragments (4 K-chunks of 32) live in registers for the entire kernel.
// Per iteration the block processes a 32-row A tile (two 16-row WMMA tiles per wave,
// two independent accumulator chains), software-pipelined: next tile's global loads
// are issued before the WMMA burst and committed to LDS after the bottom barrier.

__global__ void __launch_bounds__(256)
k_gemm_wmma(const _Float16* __restrict__ A16, const _Float16* __restrict__ W16,
            float* __restrict__ C, int M) {
    __shared__ _Float16 Alds[32 * 128];   // 8 KB

    const int t    = threadIdx.x;
    const int lane = t & 31;
    const int wave = t >> 5;       // 0..7 -> N tile
    const int g    = lane >> 4;    // lane group (0: lanes 0-15, 1: lanes 16-31)
    const int ln   = lane & 15;
    const int col  = wave * 16 + ln;

    // B fragments: 32x16 f16 in this code's contiguous K order -> half h of lane holds K = 16*g + h, N = col.
    v16h bfrag[4];
#pragma unroll
    for (int c = 0; c < 4; ++c) {
        union { v16h v; _Float16 h[16]; } u;
#pragma unroll
        for (int hh = 0; hh < 16; ++hh) {
            int K = 32 * c + 16 * g + hh;
            u.h[hh] = W16[K * 128 + col];
        }
        bfrag[c] = u.v;
    }

    const int nchunks = (M + 31) >> 5;

    // Tile loader: thread t covers two b128 segments of the 32x128 f16 tile.
    auto load_tile = [&](int ch, uint4* r) {
        const uint4* ap = (const uint4*)(A16 + (size_t)ch * 32 * 128);
#pragma unroll
        for (int s = 0; s < 2; ++s) {
            uint4 v; v.x = v.y = v.z = v.w = 0u;
            int row = ch * 32 + s * 16 + (t >> 4);
            if (row < M) v = ap[s * 256 + t];
            r[s] = v;
        }
    };

    int  chunk = blockIdx.x;
    uint4 areg[2];
    if (chunk < nchunks) load_tile(chunk, areg);

    for (; chunk < nchunks; chunk += gridDim.x) {
        // commit pipelined tile to LDS (bottom barrier guarantees LDS is free)
        ((uint4*)Alds)[t]       = areg[0];
        ((uint4*)Alds)[256 + t] = areg[1];
        __syncthreads();

        // A fragments: 16-bit A 16x32 layout.
        // VGPR v: K = 16g + 2v, +1 (contiguous K order, matches B fragments).
        // Per lane-group the 8 dwords are two contiguous 16B runs -> 2x ds_load_b128.
        v16h af[2][4];
#pragma unroll
        for (int mt = 0; mt < 2; ++mt) {
#pragma unroll
            for (int c = 0; c < 4; ++c) {
                union { v16h v; unsigned u32[8]; } u;
#pragma unroll
                for (int r = 0; r < 8; ++r) {
                    int kb = 32 * c + 16 * g + 2 * r;
                    u.u32[r] = *(const unsigned*)&Alds[(mt * 16 + ln) * 128 + kb];
                }
                af[mt][c] = u.v;
            }
        }

        // issue next tile's global loads; latency hidden under the WMMA burst
        int next = chunk + gridDim.x;
        if (next < nchunks) {
            load_tile(next, areg);
            long long prow = (long long)next * 32 + (long long)gridDim.x * 32;
            if (prow + 32 <= M) __builtin_prefetch(A16 + (size_t)prow * 128, 0, 1);
        }

        // two independent accumulator chains
        v8f acc0 = {}, acc1 = {};
#pragma unroll
        for (int c = 0; c < 4; ++c) {
            acc0 = __builtin_amdgcn_wmma_f32_16x16x32_f16(
                false, af[0][c], false, bfrag[c], (short)0, acc0, false, false);
            acc1 = __builtin_amdgcn_wmma_f32_16x16x32_f16(
                false, af[1][c], false, bfrag[c], (short)0, acc1, false, false);
        }

        // C/D layout: VGPR r -> M = r + 8*g. Uniform tile guard -> immediate-offset stores.
        const int row0 = chunk * 32;
        if (row0 + 32 <= M) {
            float* cp0 = C + (size_t)(row0 + 8 * g) * 128 + col;
            float* cp1 = cp0 + (size_t)16 * 128;
#pragma unroll
            for (int r = 0; r < 8; ++r) {
                cp0[r * 128] = acc0[r];
                cp1[r * 128] = acc1[r];
            }
        } else {
#pragma unroll
            for (int r = 0; r < 8; ++r) {
                int rw0 = row0 + r + 8 * g;
                int rw1 = rw0 + 16;
                if (rw0 < M) C[(size_t)rw0 * 128 + col] = acc0[r];
                if (rw1 < M) C[(size_t)rw1 * 128 + col] = acc1[r];
            }
        }
        __syncthreads();
    }
}

// ---------------- sparse aggregate: out[dst] += hW[src] * norm ----------------
// One wave per edge: 32 lanes x float4 = 128 features. Gathers stay in the 192MB L2
// (feature matrix is 51MB); scatters use L2 atomic units.

__global__ void __launch_bounds__(256)
k_aggregate(const float4* __restrict__ hW, const int* __restrict__ src,
            const int* __restrict__ dst, const float* __restrict__ norm,
            float* __restrict__ out, int E) {
    int e = blockIdx.x * 8 + (threadIdx.x >> 5);
    if (e >= E) return;
    int    lane = threadIdx.x & 31;
    int    s = src[e], d = dst[e];
    float  nr = norm[e];
    float4 v  = hW[(size_t)s * 32 + lane];
    float* o  = out + (size_t)d * 128 + (size_t)lane * 4;
    atomicAdd(o + 0, v.x * nr);
    atomicAdd(o + 1, v.y * nr);
    atomicAdd(o + 2, v.z * nr);
    atomicAdd(o + 3, v.w * nr);
}

// ---------------- fused epilogue: self-loop + bias + ReLU + BN + f16 convert ----------------

__global__ void k_epilogue(const float* __restrict__ hW, float* __restrict__ agg,
                           _Float16* __restrict__ h16, const float* __restrict__ dinv,
                           const float* __restrict__ bias, const float* __restrict__ gma,
                           const float* __restrict__ bet, const float* __restrict__ rm,
                           const float* __restrict__ rv, long long n) {
    long long i = (long long)blockIdx.x * blockDim.x + threadIdx.x;
    if (i >= n) return;
    int       f    = (int)(i & 127);
    long long node = i >> 7;
    float di  = dinv[node];
    float val = agg[i] + hW[i] * di * di + bias[f];   // self-loop folded in (norm = dinv^2)
    val = fmaxf(val, 0.0f);
    val = (val - rm[f]) * rsqrtf(rv[f] + BN_EPS) * gma[f] + bet[f];
    agg[i] = val;                 // f32 result (consumed by classifier after layer 3)
    h16[i] = (_Float16)val;       // f16 input for next layer's WMMA GEMM
}

// ---------------- classifier: out[N x 2] = h @ Wc + bc ----------------

__global__ void k_classifier(const float* __restrict__ h, const float* __restrict__ Wc,
                             const float* __restrict__ bc, float* __restrict__ out, int N) {
    int i = blockIdx.x * blockDim.x + threadIdx.x;
    if (i >= N) return;
    float a0 = bc[0], a1 = bc[1];
    const float4* row = (const float4*)(h + (size_t)i * 128);
#pragma unroll
    for (int q = 0; q < 32; ++q) {
        float4 v = row[q];
        int f = q * 4;
        a0 += v.x * Wc[(f + 0) * 2]     + v.y * Wc[(f + 1) * 2]
            + v.z * Wc[(f + 2) * 2]     + v.w * Wc[(f + 3) * 2];
        a1 += v.x * Wc[(f + 0) * 2 + 1] + v.y * Wc[(f + 1) * 2 + 1]
            + v.z * Wc[(f + 2) * 2 + 1] + v.w * Wc[(f + 3) * 2 + 1];
    }
    out[(size_t)i * 2 + 0] = a0;
    out[(size_t)i * 2 + 1] = a1;
}

// ---------------- launch ----------------

extern "C" void kernel_launch(void* const* d_in, const int* in_sizes, int n_in,
                              void* d_out, int out_size, void* d_ws, size_t ws_size,
                              hipStream_t stream) {
    (void)n_in; (void)out_size; (void)ws_size;

    const int N = in_sizes[0] / 128;   // x: [N,128]
    const int E = in_sizes[1] / 2;     // edge_index: [2,E]

    const float* x   = (const float*)d_in[0];
    const int*   ei  = (const int*)d_in[1];
    const int*   src = ei;
    const int*   dst = ei + E;

    const float* W[3]  = { (const float*)d_in[2],  (const float*)d_in[4],  (const float*)d_in[6]  };
    const float* b[3]  = { (const float*)d_in[3],  (const float*)d_in[5],  (const float*)d_in[7]  };
    const float* gm[3] = { (const float*)d_in[8],  (const float*)d_in[12], (const float*)d_in[16] };
    const float* be[3] = { (const float*)d_in[9],  (const float*)d_in[13], (const float*)d_in[17] };
    const float* rm[3] = { (const float*)d_in[10], (const float*)d_in[14], (const float*)d_in[18] };
    const float* rv[3] = { (const float*)d_in[11], (const float*)d_in[15], (const float*)d_in[19] };
    const float* Wc = (const float*)d_in[20];
    const float* bc = (const float*)d_in[21];
    float*       out = (float*)d_out;

    // Workspace carve-out (256B aligned): dinv[N], norm[E], bufA[N*128] f32,
    // bufB[N*128] f32, h16[N*128] f16, W16[128*128] f16. ~135 MB total.
    char*  ws  = (char*)d_ws;
    size_t off = 0;
    auto carve = [&](size_t bytes) -> void* {
        off = (off + 255) & ~(size_t)255;
        void* p = ws + off;
        off += bytes;
        return p;
    };
    const long long NH = (long long)N * 128;
    float*    dinv  = (float*)carve((size_t)N * 4);
    float*    norm  = (float*)carve((size_t)E * 4);
    float*    bufA  = (float*)carve((size_t)NH * 4);   // GEMM product h@W
    float*    bufB  = (float*)carve((size_t)NH * 4);   // aggregate / layer output
    _Float16* h16   = (_Float16*)carve((size_t)NH * 2);
    _Float16* W16   = (_Float16*)carve((size_t)128 * 128 * 2);

    const int TB = 256;

    // 1) symmetric GCN normalization
    k_fill1<<<cdiv_i(N, TB), TB, 0, stream>>>(dinv, N);
    k_deg<<<cdiv_i(E, TB), TB, 0, stream>>>(dst, dinv, E);
    k_rsqrt_inplace<<<cdiv_i(N, TB), TB, 0, stream>>>(dinv, N);
    k_norm<<<cdiv_i(E, TB), TB, 0, stream>>>(src, dst, dinv, norm, E);

    // 2) features -> f16 for WMMA
    k_cvt_f32_f16<<<cdiv_i(NH, TB), TB, 0, stream>>>(x, h16, NH);

    // 3) three GCN layers
    for (int l = 0; l < 3; ++l) {
        k_cvt_f32_f16<<<cdiv_i(128 * 128, TB), TB, 0, stream>>>(W[l], W16, 128 * 128);
        k_gemm_wmma<<<1024, TB, 0, stream>>>(h16, W16, bufA, N);
        k_zero4<<<2048, TB, 0, stream>>>((float4*)bufB, NH / 4);
        k_aggregate<<<cdiv_i(E, 8), TB, 0, stream>>>((const float4*)bufA, src, dst, norm, bufB, E);
        k_epilogue<<<cdiv_i(NH, TB), TB, 0, stream>>>(bufA, bufB, h16, dinv,
                                                      b[l], gm[l], be[l], rm[l], rv[l], NH);
    }

    // 4) classifier
    k_classifier<<<cdiv_i(N, TB), TB, 0, stream>>>(bufB, Wc, bc, out, N);
}
